// LorentzBlock_3384434229840
// MI455X (gfx1250) — matrix operand order repaired and checked by reading, not compiled, not run on hardware
//
#include <hip/hip_runtime.h>

typedef _Float16 h8  __attribute__((ext_vector_type(8)));
typedef _Float16 h16 __attribute__((ext_vector_type(16)));
typedef float    f8  __attribute__((ext_vector_type(8)));

__device__ inline h16 join8(h8 lo, h8 hi) {
  h16 r;
#pragma unroll
  for (int i = 0; i < 8; i++) { r[i] = lo[i]; r[i + 8] = hi[i]; }
  return r;
}

__device__ inline float gelu_f(float x) {
  return 0.5f * x * (1.0f + erff(x * 0.7071067811865476f));
}

// ---------------- elementwise helpers ----------------

__global__ __launch_bounds__(256) void cvt_f32_f16(const float* __restrict__ s,
                                                   _Float16* __restrict__ d, int n) {
  int i = blockIdx.x * 256 + threadIdx.x;
  if (i < n) d[i] = (_Float16)s[i];
}

__global__ __launch_bounds__(256) void mask_to_f32(const unsigned char* __restrict__ s,
                                                   float* __restrict__ d, int n) {
  int i = blockIdx.x * 256 + threadIdx.x;
  if (i < n) d[i] = s[i] ? 1.0f : 0.0f;
}

// V [BL, D] (f16) -> Vt [B,H,DH,L] (f16)
__global__ __launch_bounds__(256) void vtrans_kernel(const _Float16* __restrict__ V,
                                                     _Float16* __restrict__ Vt,
                                                     int Bn, int Ln, int Dn, int Hn) {
  size_t i = (size_t)blockIdx.x * 256 + threadIdx.x;
  size_t total = (size_t)Bn * Ln * Dn;
  if (i >= total) return;
  int c = (int)(i % Dn);
  size_t r = i / Dn;
  int l = (int)(r % Ln);
  int b = (int)(r / Ln);
  int h = c >> 6, d = c & 63;
  Vt[((size_t)(b * Hn + h) * 64 + d) * Ln + l] = V[i];
}

// ---------------- Minkowski LayerNorm (blend 0.5) ----------------
// one block of 256 threads per row, D = 768 (3 elems/thread)
__global__ __launch_bounds__(256) void mink_norm_kernel(
    const float* __restrict__ x, const float* __restrict__ m,
    const float* __restrict__ w, const float* __restrict__ b,
    _Float16* __restrict__ oa, _Float16* __restrict__ ot, _Float16* __restrict__ os,
    int D) {
  __shared__ float red[256];
  int row = blockIdx.x, tid = threadIdx.x;
  const float* xr = x + (size_t)row * D;
  float v[3];
  float s = 0.f;
#pragma unroll
  for (int i = 0; i < 3; i++) { v[i] = xr[tid + 256 * i]; s += v[i]; }
  red[tid] = s; __syncthreads();
  for (int t = 128; t > 0; t >>= 1) { if (tid < t) red[tid] += red[tid + t]; __syncthreads(); }
  float mean = red[0] / (float)D; __syncthreads();
  float vs = 0.f, es = 0.f;
#pragma unroll
  for (int i = 0; i < 3; i++) {
    int e = tid + 256 * i;
    float xc = v[i] - mean, sq = xc * xc;
    vs += sq;
    es += sq * (1.f - 2.f * m[e]);
  }
  red[tid] = vs; __syncthreads();
  for (int t = 128; t > 0; t >>= 1) { if (tid < t) red[tid] += red[tid + t]; __syncthreads(); }
  float var = red[0] / (float)D; __syncthreads();
  red[tid] = es; __syncthreads();
  for (int t = 128; t > 0; t >>= 1) { if (tid < t) red[tid] += red[tid + t]; __syncthreads(); }
  float eta = red[0];
  float rstd = 1.f / sqrtf(var + 1e-5f);
  float reta = 1.f / sqrtf(fabsf(eta) + 1e-5f);
#pragma unroll
  for (int i = 0; i < 3; i++) {
    int e = tid + 256 * i;
    float xc = v[i] - mean;
    float xn = 0.5f * xc * rstd + 0.5f * xc * reta;
    float y = w[e] * xn + b[e];
    size_t idx = (size_t)row * D + e;
    float md = m[e];
    oa[idx] = (_Float16)y;
    if (ot) ot[idx] = (_Float16)(y * md);
    if (os) os[idx] = (_Float16)(y * (1.f - md));
  }
}

// ---------------- Qeff: fold Lorentz correction + 1/sqrt(DH) into Q ----------------
// one 64-thread block per (b, l, h) head-row; DH = 64
__global__ __launch_bounds__(64) void qeff_kernel(const float* __restrict__ Q,
                                                  const float* __restrict__ m,
                                                  _Float16* __restrict__ Qe,
                                                  int H, int D) {
  __shared__ float r1[64], r2[64];
  int rowh = blockIdx.x;
  int row = rowh / H, h = rowh % H;
  int d = threadIdx.x;
  size_t base = (size_t)row * D + h * 64;
  float q = Q[base + d];
  float md = m[h * 64 + d];
  float qt = q * md;
  r1[d] = q * q; r2[d] = qt * qt; __syncthreads();
  for (int t = 32; t > 0; t >>= 1) {
    if (d < t) { r1[d] += r1[d + t]; r2[d] += r2[d + t]; }
    __syncthreads();
  }
  float qn = sqrtf(r1[0]);
  float qtn = sqrtf(r2[0]);
  float sf = (qtn > 1e-6f) ? qn / fmaxf(qtn, 1e-8f) : 0.f;
  // 2*alpha = 0.5 ; 1/SCALE = 0.125
  Qe[base + d] = (_Float16)(q * (1.f - 0.5f * sf * md) * 0.125f);
}

// ---------------- WMMA GEMM: C[M,N] = act(A[M,K] @ W[N,K]^T) ----------------
// one 32x64 output block per wave (2 M-frags x 4 N-frags, 8 accumulators):
// per K-step of 32: 4 A-loads + 8 B-loads -> 8 WMMAs (1.5 b128 loads / WMMA)
__global__ __launch_bounds__(256) void wmma_gemm_kernel(
    const _Float16* __restrict__ A, const _Float16* __restrict__ W,
    int M, int N, int K,
    float* __restrict__ o32, _Float16* __restrict__ o16,
    const float* __restrict__ res, float oscale, int act) {
  int lane = threadIdx.x & 31, wave = threadIdx.x >> 5;
  int tile = blockIdx.x * 8 + wave;
  int tn = N >> 6;                       // 64-col tiles
  int tile_m = tile / tn, tile_n = tile % tn;
  if (tile_m >= (M >> 5)) return;        // 32-row tiles
  int ln = lane & 15, hf = lane >> 4;
  const _Float16* arow0 = A + (size_t)(tile_m * 32 + ln) * K;
  const _Float16* arow1 = arow0 + (size_t)16 * K;
  const _Float16* bbase = W + (size_t)(tile_n * 64 + ln) * K;
  f8 acc[2][4] = {};
  for (int kk = 0; kk < K; kk += 32) {
    h16 a0 = join8(*(const h8*)(arow0 + kk + 8 * hf),
                   *(const h8*)(arow0 + kk + 16 + 8 * hf));
    h16 a1 = join8(*(const h8*)(arow1 + kk + 8 * hf),
                   *(const h8*)(arow1 + kk + 16 + 8 * hf));
#pragma unroll
    for (int ni = 0; ni < 4; ni++) {
      const _Float16* br = bbase + (size_t)(16 * ni) * K + kk + 8 * hf;
      h16 b = join8(*(const h8*)br, *(const h8*)(br + 16));
      acc[0][ni] = __builtin_amdgcn_wmma_f32_16x16x32_f16(false, a0, false, b, (short)0,
                                                          acc[0][ni], false, false);
      acc[1][ni] = __builtin_amdgcn_wmma_f32_16x16x32_f16(false, a1, false, b, (short)0,
                                                          acc[1][ni], false, false);
    }
  }
#pragma unroll
  for (int mi = 0; mi < 2; mi++) {
#pragma unroll
    for (int ni = 0; ni < 4; ni++) {
      int col = tile_n * 64 + ni * 16 + ln;
#pragma unroll
      for (int v = 0; v < 8; v++) {
        int row = tile_m * 32 + mi * 16 + v + 8 * hf;
        float xv = acc[mi][ni][v];
        if (act == 1) xv = gelu_f(xv);
        else if (act == 2) xv = xv / (1.f + expf(-xv));
        size_t idx = (size_t)row * N + col;
        if (o16) o16[idx] = (_Float16)xv;
        if (o32) o32[idx] = (res ? res[idx] : 0.f) + oscale * xv;
      }
    }
  }
}

// ---------------- flash attention: one q-tile (16 rows) per wave ----------------
__global__ __launch_bounds__(256) void attn_kernel(
    const _Float16* __restrict__ Qe, const _Float16* __restrict__ Kx,
    const _Float16* __restrict__ Vt, _Float16* __restrict__ O,
    int Bn, int Hn, int Ln, int Dn) {
  __shared__ _Float16 plds[8][16][32];
  int lane = threadIdx.x & 31, wave = threadIdx.x >> 5;
  int qt_per = Ln >> 4;
  int g = blockIdx.x * 8 + wave;
  if (g >= Bn * Hn * qt_per) return;
  int qt = g % qt_per;
  int t2 = g / qt_per;
  int hh = t2 % Hn, bb = t2 / Hn;
  int qbase = qt * 16;
  int ln = lane & 15, hf = lane >> 4;

  const _Float16* qrow = Qe + ((size_t)bb * Ln + qbase + ln) * Dn + hh * 64;
  h16 aq0 = join8(*(const h8*)(qrow + 8 * hf), *(const h8*)(qrow + 16 + 8 * hf));
  h16 aq1 = join8(*(const h8*)(qrow + 32 + 8 * hf), *(const h8*)(qrow + 48 + 8 * hf));

  f8 o0 = {}, o1 = {}, o2 = {}, o3 = {};
  float mi[8], li[8];
#pragma unroll
  for (int v = 0; v < 8; v++) { mi[v] = -1e30f; li[v] = 0.f; }

  for (int kbase = 0; kbase <= qbase + 15; kbase += 32) {
    const _Float16* kr0 = Kx + ((size_t)bb * Ln + kbase + ln) * Dn + hh * 64;
    const _Float16* kr1 = kr0 + (size_t)16 * Dn;
    f8 s0 = {}, s1 = {};
    {
      h16 b0 = join8(*(const h8*)(kr0 + 8 * hf), *(const h8*)(kr0 + 8 * hf + 16));
      s0 = __builtin_amdgcn_wmma_f32_16x16x32_f16(false, aq0, false, b0, (short)0, s0, false, false);
      h16 b1 = join8(*(const h8*)(kr1 + 8 * hf), *(const h8*)(kr1 + 8 * hf + 16));
      s1 = __builtin_amdgcn_wmma_f32_16x16x32_f16(false, aq0, false, b1, (short)0, s1, false, false);
      h16 b2 = join8(*(const h8*)(kr0 + 32 + 8 * hf), *(const h8*)(kr0 + 32 + 8 * hf + 16));
      s0 = __builtin_amdgcn_wmma_f32_16x16x32_f16(false, aq1, false, b2, (short)0, s0, false, false);
      h16 b3 = join8(*(const h8*)(kr1 + 32 + 8 * hf), *(const h8*)(kr1 + 32 + 8 * hf + 16));
      s1 = __builtin_amdgcn_wmma_f32_16x16x32_f16(false, aq1, false, b3, (short)0, s1, false, false);
    }
    float p0a[8], p1a[8];
#pragma unroll
    for (int v = 0; v < 8; v++) {
      int qrw = qbase + v + 8 * hf;
      float sv0 = s0[v] + (((kbase + ln) <= qrw) ? 0.f : -1e9f);
      float sv1 = s1[v] + (((kbase + 16 + ln) <= qrw) ? 0.f : -1e9f);
      float r = fmaxf(sv0, sv1);
      r = fmaxf(r, __shfl_xor(r, 1, 32));
      r = fmaxf(r, __shfl_xor(r, 2, 32));
      r = fmaxf(r, __shfl_xor(r, 4, 32));
      r = fmaxf(r, __shfl_xor(r, 8, 32));
      float mnew = fmaxf(mi[v], r);
      float sc = expf(mi[v] - mnew);
      float p0 = expf(sv0 - mnew);
      float p1 = expf(sv1 - mnew);
      float rs = p0 + p1;
      rs += __shfl_xor(rs, 1, 32);
      rs += __shfl_xor(rs, 2, 32);
      rs += __shfl_xor(rs, 4, 32);
      rs += __shfl_xor(rs, 8, 32);
      li[v] = li[v] * sc + rs;
      mi[v] = mnew;
      o0[v] *= sc; o1[v] *= sc; o2[v] *= sc; o3[v] *= sc;
      p0a[v] = p0; p1a[v] = p1;
    }
    // re-layout P: C-fragment -> LDS -> A-fragment (f16)
#pragma unroll
    for (int v = 0; v < 8; v++) {
      plds[wave][v + 8 * hf][ln] = (_Float16)p0a[v];
      plds[wave][v + 8 * hf][ln + 16] = (_Float16)p1a[v];
    }
    asm volatile("s_wait_dscnt 0" ::: "memory");
    h16 pa = join8(*(const h8*)&plds[wave][ln][8 * hf],
                   *(const h8*)&plds[wave][ln][16 + 8 * hf]);
    // P (16x32) @ V (32x64): Vt rows are contiguous in key index
    const _Float16* vr = Vt + ((size_t)(bb * Hn + hh) * 64 + ln) * Ln + kbase + 8 * hf;
    o0 = __builtin_amdgcn_wmma_f32_16x16x32_f16(false, pa, false,
        join8(*(const h8*)vr, *(const h8*)(vr + 16)), (short)0, o0, false, false);
    const _Float16* v1 = vr + (size_t)16 * Ln;
    o1 = __builtin_amdgcn_wmma_f32_16x16x32_f16(false, pa, false,
        join8(*(const h8*)v1, *(const h8*)(v1 + 16)), (short)0, o1, false, false);
    const _Float16* v2 = vr + (size_t)32 * Ln;
    o2 = __builtin_amdgcn_wmma_f32_16x16x32_f16(false, pa, false,
        join8(*(const h8*)v2, *(const h8*)(v2 + 16)), (short)0, o2, false, false);
    const _Float16* v3 = vr + (size_t)48 * Ln;
    o3 = __builtin_amdgcn_wmma_f32_16x16x32_f16(false, pa, false,
        join8(*(const h8*)v3, *(const h8*)(v3 + 16)), (short)0, o3, false, false);
  }
#pragma unroll
  for (int v = 0; v < 8; v++) {
    float inv = 1.f / li[v];
    size_t obase = ((size_t)bb * Ln + qbase + v + 8 * hf) * Dn + hh * 64;
    O[obase + ln]      = (_Float16)(o0[v] * inv);
    O[obase + 16 + ln] = (_Float16)(o1[v] * inv);
    O[obase + 32 + ln] = (_Float16)(o2[v] * inv);
    O[obase + 48 + ln] = (_Float16)(o3[v] * inv);
  }
}

// ---------------- host orchestration ----------------

extern "C" void kernel_launch(void* const* d_in, const int* in_sizes, int n_in,
                              void* d_out, int out_size, void* d_ws, size_t ws_size,
                              hipStream_t stream) {
  (void)in_sizes; (void)n_in; (void)out_size; (void)ws_size;
  const int Bn = 2, Ln = 2048, Dn = 768, Hn = 12, BL = Bn * Ln, DFF = 3072, DHF = 1536;

  const float* x  = (const float*)d_in[0];
  const unsigned char* tmask = (const unsigned char*)d_in[2];
  const float* Wq = (const float*)d_in[3];
  const float* Wk = (const float*)d_in[4];
  const float* Wv = (const float*)d_in[5];
  const float* Wo = (const float*)d_in[6];
  const float* w1 = (const float*)d_in[7];
  const float* w2 = (const float*)d_in[8];
  const float* w1t = (const float*)d_in[9];
  const float* w2t = (const float*)d_in[10];
  const float* w1s = (const float*)d_in[11];
  const float* w2s = (const float*)d_in[12];
  const float* n1w = (const float*)d_in[13];
  const float* n1b = (const float*)d_in[14];
  const float* n2w = (const float*)d_in[15];
  const float* n2b = (const float*)d_in[16];
  float* out = (float*)d_out;

  char* base = (char*)d_ws;
  size_t off = 0;
  auto alloc = [&](size_t bytes) -> char* {
    off = (off + 255) & ~(size_t)255;
    char* p = base + off;
    off += bytes;
    return p;
  };

  _Float16* wq16  = (_Float16*)alloc((size_t)Dn * Dn * 2);
  _Float16* wk16  = (_Float16*)alloc((size_t)Dn * Dn * 2);
  _Float16* wv16  = (_Float16*)alloc((size_t)Dn * Dn * 2);
  _Float16* wo16  = (_Float16*)alloc((size_t)Dn * Dn * 2);
  _Float16* w116  = (_Float16*)alloc((size_t)DFF * Dn * 2);
  _Float16* w216  = (_Float16*)alloc((size_t)Dn * DFF * 2);
  _Float16* w1t16 = (_Float16*)alloc((size_t)DHF * Dn * 2);
  _Float16* w2t16 = (_Float16*)alloc((size_t)Dn * DHF * 2);
  _Float16* w1s16 = (_Float16*)alloc((size_t)DHF * Dn * 2);
  _Float16* w2s16 = (_Float16*)alloc((size_t)Dn * DHF * 2);
  float*    mf    = (float*)alloc((size_t)Dn * 4);
  _Float16* xn1   = (_Float16*)alloc((size_t)BL * Dn * 2);
  float*    Qf    = (float*)alloc((size_t)BL * Dn * 4);
  _Float16* K16   = (_Float16*)alloc((size_t)BL * Dn * 2);
  _Float16* V16   = (_Float16*)alloc((size_t)BL * Dn * 2);
  _Float16* Vt16  = (_Float16*)alloc((size_t)BL * Dn * 2);
  _Float16* Qe16  = (_Float16*)alloc((size_t)BL * Dn * 2);
  _Float16* at16  = (_Float16*)alloc((size_t)BL * Dn * 2);
  float*    x1    = (float*)alloc((size_t)BL * Dn * 4);
  _Float16* xn2   = (_Float16*)alloc((size_t)BL * Dn * 2);
  _Float16* xn2m  = (_Float16*)alloc((size_t)BL * Dn * 2);
  _Float16* xn2s  = (_Float16*)alloc((size_t)BL * Dn * 2);
  _Float16* g1    = (_Float16*)alloc((size_t)BL * DFF * 2);
  _Float16* t1    = (_Float16*)alloc((size_t)BL * DHF * 2);
  _Float16* s1    = (_Float16*)alloc((size_t)BL * DHF * 2);

  auto cvt = [&](const float* s, _Float16* d, int n) {
    cvt_f32_f16<<<(n + 255) / 256, 256, 0, stream>>>(s, d, n);
  };
  cvt(Wq, wq16, Dn * Dn);   cvt(Wk, wk16, Dn * Dn);
  cvt(Wv, wv16, Dn * Dn);   cvt(Wo, wo16, Dn * Dn);
  cvt(w1, w116, DFF * Dn);  cvt(w2, w216, Dn * DFF);
  cvt(w1t, w1t16, DHF * Dn); cvt(w2t, w2t16, Dn * DHF);
  cvt(w1s, w1s16, DHF * Dn); cvt(w2s, w2s16, Dn * DHF);
  mask_to_f32<<<3, 256, 0, stream>>>(tmask, mf, Dn);

  auto gemm = [&](const _Float16* A, const _Float16* Wm, int M, int N, int K,
                  float* o32, _Float16* o16, const float* res, float sc, int act) {
    int tiles = (M / 32) * (N / 64);
    wmma_gemm_kernel<<<(tiles + 7) / 8, 256, 0, stream>>>(A, Wm, M, N, K, o32, o16, res, sc, act);
  };

  // norm1 -> QKV
  mink_norm_kernel<<<BL, 256, 0, stream>>>(x, mf, n1w, n1b, xn1, nullptr, nullptr, Dn);
  gemm(xn1, wq16, BL, Dn, Dn, Qf, nullptr, nullptr, 1.f, 0);
  gemm(xn1, wk16, BL, Dn, Dn, nullptr, K16, nullptr, 1.f, 0);
  gemm(xn1, wv16, BL, Dn, Dn, nullptr, V16, nullptr, 1.f, 0);

  // attention
  qeff_kernel<<<BL * Hn, 64, 0, stream>>>(Qf, mf, Qe16, Hn, Dn);
  vtrans_kernel<<<(BL * Dn) / 256, 256, 0, stream>>>(V16, Vt16, Bn, Ln, Dn, Hn);
  attn_kernel<<<(Bn * Hn * (Ln / 16)) / 8, 256, 0, stream>>>(Qe16, K16, Vt16, at16, Bn, Hn, Ln, Dn);
  gemm(at16, wo16, BL, Dn, Dn, x1, nullptr, x, 1.f, 0);  // x1 = x + attn @ Wo^T

  // norm2 -> FFN
  mink_norm_kernel<<<BL, 256, 0, stream>>>(x1, mf, n2w, n2b, xn2, xn2m, xn2s, Dn);
  gemm(xn2,  w116,  BL, DFF, Dn, nullptr, g1, nullptr, 1.f, 1);  // gelu
  gemm(xn2m, w1t16, BL, DHF, Dn, nullptr, t1, nullptr, 1.f, 2);  // silu
  gemm(xn2s, w1s16, BL, DHF, Dn, nullptr, s1, nullptr, 1.f, 1);  // gelu

  // out = x1 + 0.5*std + 0.5*ht + 0.5*hs (accumulated in-place in d_out)
  gemm(g1, w216,  BL, Dn, DFF, out, nullptr, x1,  0.5f, 0);
  gemm(t1, w2t16, BL, Dn, DHF, out, nullptr, out, 0.5f, 0);
  gemm(s1, w2s16, BL, Dn, DHF, out, nullptr, out, 0.5f, 0);
}
